// mainfold_attention_71768903516351
// MI455X (gfx1250) — matrix operand order repaired and checked by reading, not compiled, not run on hardware
//
#include <hip/hip_runtime.h>

// ---------------------------------------------------------------------------
// Problem constants (match reference)
// ---------------------------------------------------------------------------
#define BDIM 64
#define CDIM 1024
#define HW   100
#define NT   100       // image tokens
#define MT   20        // text tokens
#define P_VI 200       // joint vis+ir points
#define P_IT 120       // joint img+text points
#define SQC  32.0f     // sqrt(1024)
#define BIGV 1.0e6f

// ---------------------------------------------------------------------------
// WMMA types (CDNA5 / gfx1250, wave32)
// ---------------------------------------------------------------------------
typedef __attribute__((ext_vector_type(16))) __bf16 v16bf;
typedef __attribute__((ext_vector_type(8)))  float  v8f;

union FragU { unsigned int u[8]; v16bf v; };

__device__ __forceinline__ unsigned short f32_to_bf16(float f) {
  unsigned int u = __float_as_uint(f);
  return (unsigned short)((u + 0x7FFFu + ((u >> 16) & 1u)) >> 16);
}
__device__ __forceinline__ unsigned int pack_bf16(float v0, float v1) {
  return (unsigned int)f32_to_bf16(v0) | ((unsigned int)f32_to_bf16(v1) << 16);
}
__device__ __forceinline__ float sigmoidf(float x) { return 1.f / (1.f + expf(-x)); }

// ---------------------------------------------------------------------------
// Generic batched GEMM:  C[b,m,n] (+)= alpha * sum_k A[b,m,k]*B[b,k,n] (+bias)
// Stride-parameterized (transposes + broadcast weights).  bf16 operands in
// LDS, f32 accumulation via v_wmma_f32_16x16x32_bf16.
// Block tile 64x128, 256 threads = 8 waves in 2x4 grid; each wave owns a
// 32x32 slab = 2 A-frags x 2 B-frags -> 4 WMMAs per K-step (2x frag reuse).
// Interior blocks take a branch-free unconditional-load path; edge blocks use
// clamp+select (no divergent exec branches in staging).
// ---------------------------------------------------------------------------
#define GBM 64
#define GBN 128
#define GKT 32

__device__ __forceinline__ void wmma_4x(
    const unsigned short (&As)[GBM][GKT + 2], const unsigned short (&Bs)[GBN][GKT + 2],
    int wm, int wn, int half, int mr,
    v8f& a00, v8f& a01, v8f& a10, v8f& a11)
{
  FragU fa0, fa1, fb0, fb1;
  const int ar0 = wm * 32 + mr, ar1 = ar0 + 16;
  #pragma unroll
  for (int v = 0; v < 8; ++v) {
    int kk = ((v < 4) ? 0 : 16) + half * 8 + ((v & 3) << 1);
    fa0.u[v] = *(const unsigned int*)&As[ar0][kk];
    fa1.u[v] = *(const unsigned int*)&As[ar1][kk];
  }
  const int br0 = wn * 32 + mr, br1 = br0 + 16;
  #pragma unroll
  for (int v = 0; v < 8; ++v) {
    int kk = ((v < 4) ? 0 : 16) + half * 8 + ((v & 3) << 1);
    fb0.u[v] = *(const unsigned int*)&Bs[br0][kk];
    fb1.u[v] = *(const unsigned int*)&Bs[br1][kk];
  }
  a00 = __builtin_amdgcn_wmma_f32_16x16x32_bf16(false, fa0.v, false, fb0.v, (short)0, a00, false, false);
  a01 = __builtin_amdgcn_wmma_f32_16x16x32_bf16(false, fa0.v, false, fb1.v, (short)0, a01, false, false);
  a10 = __builtin_amdgcn_wmma_f32_16x16x32_bf16(false, fa1.v, false, fb0.v, (short)0, a10, false, false);
  a11 = __builtin_amdgcn_wmma_f32_16x16x32_bf16(false, fa1.v, false, fb1.v, (short)0, a11, false, false);
}

__launch_bounds__(256)
__global__ void gemm_wmma_bf16(
    const float* __restrict__ A, const float* __restrict__ Bm, float* __restrict__ Cm,
    int M, int N, int K,
    long sAb, long sAm, long sAk,
    long sBb, long sBk, long sBn,
    long sCb, long sCm,
    const float* __restrict__ bias, int biasMode,        // 0 none, 1 per-n, 2 per-m
    const float* __restrict__ alphaPtr, int alphaIdx, float alphaConst,
    int flags)                                           // bit0: accumulate, bit1: relu
{
  __shared__ __align__(16) unsigned short As[GBM][GKT + 2];
  __shared__ __align__(16) unsigned short Bs[GBN][GKT + 2];

  const int b    = blockIdx.z;
  const int m0   = blockIdx.y * GBM;
  const int n0   = blockIdx.x * GBN;
  const int tid  = threadIdx.x;
  const int lane = tid & 31;
  const int wave = tid >> 5;
  const int wm   = wave >> 2;   // 0..1 : 32-row strip
  const int wn   = wave & 3;    // 0..3 : 32-col strip
  const int half = lane >> 4;
  const int mr   = lane & 15;

  const float* Ab = A  + (long)b * sAb;
  const float* Bb = Bm + (long)b * sBb;

  float alpha = alphaConst;
  if (alphaPtr) alpha *= alphaPtr[alphaIdx];

  v8f a00 = {}, a01 = {}, a10 = {}, a11 = {};

  const bool interiorMN = (m0 + GBM <= M) && (n0 + GBN <= N);
  const bool interior   = interiorMN && ((K & (GKT - 1)) == 0);

  if (interior) {
    // ---------------- fast path: no bounds logic at all ----------------
    for (int k0 = 0; k0 < K; k0 += GKT) {
      #pragma unroll
      for (int t = 0; t < 4; ++t) {            // A: 64x32 = 1024 bf16-pairs
        int e = tid + t * 256;
        int i = e >> 4, kp = (e & 15) << 1;
        long base = (long)(m0 + i) * sAm + (long)(k0 + kp) * sAk;
        *(unsigned int*)&As[i][kp] = pack_bf16(Ab[base], Ab[base + sAk]);
      }
      #pragma unroll
      for (int t = 0; t < 8; ++t) {            // B: 128x32 = 2048 bf16-pairs
        int e = tid + t * 256;
        int nn = e >> 4, kp = (e & 15) << 1;
        long base = (long)(k0 + kp) * sBk + (long)(n0 + nn) * sBn;
        *(unsigned int*)&Bs[nn][kp] = pack_bf16(Bb[base], Bb[base + sBk]);
      }
      if (k0 + GKT < K) {                      // global_prefetch next strips
        __builtin_prefetch(&Ab[(long)(m0 + (tid >> 2)) * sAm + (long)(k0 + GKT) * sAk], 0, 3);
        __builtin_prefetch(&Bb[(long)(k0 + GKT) * sBk + (long)(n0 + (tid >> 1)) * sBn], 0, 3);
      }
      __syncthreads();
      wmma_4x(As, Bs, wm, wn, half, mr, a00, a01, a10, a11);
      __syncthreads();
    }
  } else {
    // ---------------- edge path: clamp + select (branch-free) ----------------
    for (int k0 = 0; k0 < K; k0 += GKT) {
      #pragma unroll
      for (int t = 0; t < 4; ++t) {
        int e = tid + t * 256;
        int i = e >> 4, kp = (e & 15) << 1;
        int gm = m0 + i, gk = k0 + kp;
        int cm = gm < M ? gm : M - 1;
        int c0 = gk < K ? gk : K - 1;
        int c1 = gk + 1 < K ? gk + 1 : K - 1;
        float v0 = Ab[(long)cm * sAm + (long)c0 * sAk];
        float v1 = Ab[(long)cm * sAm + (long)c1 * sAk];
        v0 = (gm < M && gk < K) ? v0 : 0.f;
        v1 = (gm < M && gk + 1 < K) ? v1 : 0.f;
        *(unsigned int*)&As[i][kp] = pack_bf16(v0, v1);
      }
      #pragma unroll
      for (int t = 0; t < 8; ++t) {
        int e = tid + t * 256;
        int nn = e >> 4, kp = (e & 15) << 1;
        int gn = n0 + nn, gk = k0 + kp;
        int cn = gn < N ? gn : N - 1;
        int c0 = gk < K ? gk : K - 1;
        int c1 = gk + 1 < K ? gk + 1 : K - 1;
        float v0 = Bb[(long)c0 * sBk + (long)cn * sBn];
        float v1 = Bb[(long)c1 * sBk + (long)cn * sBn];
        v0 = (gn < N && gk < K) ? v0 : 0.f;
        v1 = (gn < N && gk + 1 < K) ? v1 : 0.f;
        *(unsigned int*)&Bs[nn][kp] = pack_bf16(v0, v1);
      }
      __syncthreads();
      wmma_4x(As, Bs, wm, wn, half, mr, a00, a01, a10, a11);
      __syncthreads();
    }
  }

  // ---- epilogue: C/D layout -> lane(mr)=col, vgpr v + 8*half = row ----
  float* Cb = Cm + (long)b * sCb;
  #pragma unroll
  for (int i = 0; i < 2; ++i) {
    #pragma unroll
    for (int j = 0; j < 2; ++j) {
      const v8f& acc = (i == 0) ? (j == 0 ? a00 : a01) : (j == 0 ? a10 : a11);
      int gn = n0 + wn * 32 + j * 16 + mr;
      if (interiorMN) {
        #pragma unroll
        for (int v = 0; v < 8; ++v) {
          int gm = m0 + wm * 32 + i * 16 + half * 8 + v;
          float val = alpha * acc[v];
          if (biasMode == 1) val += bias[gn];
          else if (biasMode == 2) val += bias[gm];
          long ci = (long)gm * sCm + gn;
          if (flags & 1) val += Cb[ci];
          if (flags & 2) val = fmaxf(val, 0.f);
          Cb[ci] = val;
        }
      } else if (gn < N) {
        #pragma unroll
        for (int v = 0; v < 8; ++v) {
          int gm = m0 + wm * 32 + i * 16 + half * 8 + v;
          if (gm >= M) continue;
          float val = alpha * acc[v];
          if (biasMode == 1) val += bias[gn];
          else if (biasMode == 2) val += bias[gm];
          long ci = (long)gm * sCm + gn;
          if (flags & 1) val += Cb[ci];
          if (flags & 2) val = fmaxf(val, 0.f);
          Cb[ci] = val;
        }
      }
    }
  }
}

// ---------------------------------------------------------------------------
// Geodesic kernel: whole PxP matrix resident in CDNA5 320KB LDS.
// From Gram matrix -> Euclidean D -> 8-NN graph -> Floyd-Warshall -> slice.
// One workgroup per batch sample.
// ---------------------------------------------------------------------------
__launch_bounds__(1024)
__global__ void geodesic_kernel(const float* __restrict__ G, float* __restrict__ out,
                                int P, int rowStart, int rowCount, int colStart, int colCount)
{
  extern __shared__ float Wm[];                  // P*P floats (dyn LDS, up to 160KB)
  __shared__ float diag[256];
  __shared__ float knn_d[256][8];
  __shared__ int   knn_j[256][8];

  const int b = blockIdx.x;
  const float* Gb = G + (long)b * P * P;
  const int tid = threadIdx.x, nt = blockDim.x;
  const int PP = P * P;

  for (int p = tid; p < P; p += nt) diag[p] = Gb[p * P + p];
  __syncthreads();

  // Euclidean distances, +BIG on diagonal (exclude self from kNN)
  for (int e = tid; e < PP; e += nt) {
    int i = e / P, j = e - i * P;
    float d2 = diag[i] + diag[j] - 2.f * Gb[e];
    float d = sqrtf(fmaxf(d2, 0.f));
    Wm[e] = (i == j) ? d + BIGV : d;
  }
  __syncthreads();

  // 8-NN per row (insertion top-k)
  for (int i = tid; i < P; i += nt) {
    float bd[8]; int bj[8];
    #pragma unroll
    for (int t = 0; t < 8; ++t) { bd[t] = 3.0e38f; bj[t] = 0; }
    for (int j = 0; j < P; ++j) {
      float d = Wm[i * P + j];
      if (d < bd[7]) {
        int t = 7;
        while (t > 0 && bd[t - 1] > d) { bd[t] = bd[t - 1]; bj[t] = bj[t - 1]; --t; }
        bd[t] = d; bj[t] = j;
      }
    }
    #pragma unroll
    for (int t = 0; t < 8; ++t) { knn_d[i][t] = bd[t]; knn_j[i][t] = bj[t]; }
  }
  __syncthreads();

  // adjacency: BIG everywhere, 0 diag, kNN edges, symmetrized with min
  for (int e = tid; e < PP; e += nt) {
    int i = e / P, j = e - i * P;
    Wm[e] = (i == j) ? 0.f : BIGV;
  }
  __syncthreads();
  for (int i = tid; i < P; i += nt)
    #pragma unroll
    for (int t = 0; t < 8; ++t) Wm[i * P + knn_j[i][t]] = knn_d[i][t];
  __syncthreads();
  for (int e = tid; e < PP; e += nt) {
    int i = e / P, j = e - i * P;
    if (i < j) {
      float m = fminf(Wm[i * P + j], Wm[j * P + i]);
      Wm[i * P + j] = m; Wm[j * P + i] = m;
    }
  }
  __syncthreads();

  // Floyd-Warshall (in-place; row/col k invariant at step k)
  for (int k = 0; k < P; ++k) {
    for (int e = tid; e < PP; e += nt) {
      int i = e / P, j = e - i * P;
      float w = Wm[i * P + k] + Wm[k * P + j];
      if (w < Wm[e]) Wm[e] = w;
    }
    __syncthreads();
  }

  float* ob = out + (long)b * rowCount * colCount;
  for (int e = tid; e < rowCount * colCount; e += nt) {
    int r = e / colCount, c = e - r * colCount;
    ob[e] = Wm[(rowStart + r) * P + colStart + c];
  }
}

// ---------------------------------------------------------------------------
// Per-batch min-max normalization (in place)
// ---------------------------------------------------------------------------
__launch_bounds__(256)
__global__ void minmax_norm(float* x, int perBatch)
{
  const int b = blockIdx.x, tid = threadIdx.x;
  float* xb = x + (long)b * perBatch;
  __shared__ float smn[256], smx[256];
  float mn = 3.0e38f, mx = -3.0e38f;
  for (int e = tid; e < perBatch; e += 256) {
    float v = xb[e];
    mn = fminf(mn, v); mx = fmaxf(mx, v);
  }
  smn[tid] = mn; smx[tid] = mx; __syncthreads();
  for (int s = 128; s; s >>= 1) {
    if (tid < s) { smn[tid] = fminf(smn[tid], smn[tid + s]); smx[tid] = fmaxf(smx[tid], smx[tid + s]); }
    __syncthreads();
  }
  mn = smn[0]; mx = smx[0];
  float inv = 1.f / (mx - mn + 1e-8f);
  for (int e = tid; e < perBatch; e += 256) xb[e] = (xb[e] - mn) * inv;
}

// ---------------------------------------------------------------------------
// Median via deterministic bisection on [0,1]
// state: [0]=lo [1]=hi [2]=sigma1 [3]=sigma2 [4..13]=loss accums; cnt at +16
// ---------------------------------------------------------------------------
__launch_bounds__(256)
__global__ void count_le(const float* __restrict__ x, long n,
                         const float* __restrict__ state, unsigned int* cnt)
{
  float mid = 0.5f * (state[0] + state[1]);
  unsigned int c = 0;
  for (long i = (long)blockIdx.x * blockDim.x + threadIdx.x; i < n; i += (long)gridDim.x * blockDim.x)
    if (x[i] <= mid) ++c;
  __shared__ unsigned int scc[256];
  scc[threadIdx.x] = c; __syncthreads();
  for (int s = 128; s; s >>= 1) { if (threadIdx.x < (unsigned)s) scc[threadIdx.x] += scc[threadIdx.x + s]; __syncthreads(); }
  if (threadIdx.x == 0) atomicAdd(cnt, scc[0]);
}

__global__ void bisect_update(float* state, unsigned int* cnt, long n)
{
  float mid = 0.5f * (state[0] + state[1]);
  if (2ll * (long long)(*cnt) >= n) state[1] = mid; else state[0] = mid;
  *cnt = 0;
}

__global__ void finalize_median(float* state, int slot)
{
  state[slot] = 0.5f * (state[0] + state[1]);
  state[0] = 0.f; state[1] = 1.f;
}

__global__ void init_state(float* state, unsigned int* cnt)
{
  state[0] = 0.f; state[1] = 1.f;
  for (int i = 2; i < 16; ++i) state[i] = 0.f;
  *cnt = 0u;
}

// ---------------------------------------------------------------------------
// InstanceNorm(affine)+ReLU over tokens, x layout [B, ntok, C]
// ---------------------------------------------------------------------------
__launch_bounds__(128)
__global__ void inorm_relu(float* __restrict__ x, const float* __restrict__ gamma,
                           const float* __restrict__ beta, int ntok)
{
  const long bc = blockIdx.x;
  const int c = (int)(bc % CDIM);
  float* xb = x + (bc / CDIM) * (long)ntok * CDIM + c;
  const int tid = threadIdx.x;
  __shared__ float rs[128], rq[128];
  float s = 0.f, q = 0.f;
  for (int t = tid; t < ntok; t += 128) { float v = xb[(long)t * CDIM]; s += v; q += v * v; }
  rs[tid] = s; rq[tid] = q; __syncthreads();
  for (int st = 64; st; st >>= 1) {
    if (tid < st) { rs[tid] += rs[tid + st]; rq[tid] += rq[tid + st]; }
    __syncthreads();
  }
  float mu  = rs[0] / ntok;
  float var = rq[0] / ntok - mu * mu;
  float inv = rsqrtf(var + 1e-5f);
  float g = gamma[c], be = beta[c];
  for (int t = tid; t < ntok; t += 128) {
    float v = xb[(long)t * CDIM];
    xb[(long)t * CDIM] = fmaxf((v - mu) * inv * g + be, 0.f);
  }
}

// ---------------------------------------------------------------------------
// Mgap: per-batch mean of M_v2i_learn (fix/learn computed on the fly)
// ---------------------------------------------------------------------------
__launch_bounds__(256)
__global__ void mgap_reduce(const float* __restrict__ geo, const float* __restrict__ W1,
                            const float* __restrict__ b1, const float* __restrict__ sc,
                            const float* __restrict__ state, float* __restrict__ Mgap)
{
  const int b = blockIdx.x, tid = threadIdx.x;
  const float* g = geo + (long)b * (NT * NT);
  float sig = state[2];
  float inv = 1.f / (2.f * sig * sig);
  float lam = sigmoidf(sc[0]);
  float s = 0.f;
  for (int e = tid; e < NT * NT; e += 256) {
    float d = g[e];
    float fx = expf(-d * d * inv);
    s += lam * fx + (1.f - lam) * (W1[e] * fx + b1[e]);
  }
  __shared__ float red[256];
  red[tid] = s; __syncthreads();
  for (int st = 128; st; st >>= 1) { if (tid < st) red[tid] += red[tid + st]; __syncthreads(); }
  if (tid == 0) Mgap[b] = red[0] / (float)(NT * NT);
}

__global__ void alphabeta(const float* Mgap, const float* sc, float* al, float* be)
{
  int b = threadIdx.x;
  if (b < BDIM) {
    al[b] = sigmoidf(sc[8]  * Mgap[b] + sc[9]);
    be[b] = sigmoidf(sc[10] * Mgap[b] + sc[11]);
  }
}

// ---------------------------------------------------------------------------
// Elementwise / transpose / concat helpers
// ---------------------------------------------------------------------------
__launch_bounds__(256)
__global__ void fuse_ab(const float* __restrict__ X, const float* __restrict__ Y,
                        const float* __restrict__ al, const float* __restrict__ be,
                        float* __restrict__ out, long perBatch, long n)
{
  long i = (long)blockIdx.x * 256 + threadIdx.x;
  if (i < n) { long b = i / perBatch; out[i] = al[b] * X[i] + be[b] * Y[i]; }
}

__launch_bounds__(256)
__global__ void feat_img_kernel(const float* __restrict__ vis, const float* __restrict__ ir,
                                const float* __restrict__ al, const float* __restrict__ be,
                                float* __restrict__ out)
{
  long i = (long)blockIdx.x * 256 + threadIdx.x;
  long n = (long)BDIM * NT * CDIM;
  if (i >= n) return;
  long b = i / (NT * CDIM);
  long r = i - b * (NT * CDIM);
  long t = r / CDIM, c = r - t * CDIM;
  long src = b * (long)CDIM * HW + c * HW + t;
  out[i] = al[b] * vis[src] + be[b] * ir[src];
}

__launch_bounds__(256)
__global__ void make_joint_vi(const float* __restrict__ vis, const float* __restrict__ ir,
                              float* __restrict__ out)
{
  long i = (long)blockIdx.x * 256 + threadIdx.x;
  long n = (long)BDIM * P_VI * CDIM;
  if (i >= n) return;
  long b = i / (P_VI * CDIM);
  long r = i - b * (P_VI * CDIM);
  long p = r / CDIM, c = r - p * CDIM;
  const float* src = (p < NT) ? vis : ir;
  long pp = (p < NT) ? p : p - NT;
  out[i] = src[b * (long)CDIM * HW + c * HW + pp];
}

__launch_bounds__(256)
__global__ void lan_transpose(const float* __restrict__ lan, float* __restrict__ out)
{
  long i = (long)blockIdx.x * 256 + threadIdx.x;
  long n = (long)BDIM * MT * CDIM;
  if (i >= n) return;
  long b = i / (MT * CDIM);
  long r = i - b * (MT * CDIM);
  long t = r / CDIM, c = r - t * CDIM;
  out[i] = lan[b * (long)CDIM * MT + c * MT + t];
}

__launch_bounds__(256)
__global__ void make_joint_it(const float* __restrict__ feat, const float* __restrict__ lanT,
                              float* __restrict__ out)
{
  long i = (long)blockIdx.x * 256 + threadIdx.x;
  long n = (long)BDIM * P_IT * CDIM;
  if (i >= n) return;
  long b = i / (P_IT * CDIM);
  long r = i - b * (P_IT * CDIM);
  long p = r / CDIM, c = r - p * CDIM;
  out[i] = (p < NT) ? feat[b * (long)NT * CDIM + p * CDIM + c]
                    : lanT[b * (long)MT * CDIM + (p - NT) * CDIM + c];
}

// fix = exp(-d^2/(2*sigma2^2)); write fix [B,MT,NT] and learn (transposed) [B,NT,MT]
__launch_bounds__(256)
__global__ void fix_learn_text(const float* __restrict__ geo, const float* __restrict__ W2,
                               const float* __restrict__ b2, const float* __restrict__ sc,
                               const float* __restrict__ state,
                               float* __restrict__ fixOut, float* __restrict__ preOut)
{
  long idx = (long)blockIdx.x * 256 + threadIdx.x;
  long n = (long)BDIM * MT * NT;
  if (idx >= n) return;
  long b = idx / (MT * NT);
  long r = idx - b * (MT * NT);
  long t = r / NT, i = r - t * NT;
  float sig = state[3];
  float inv = 1.f / (2.f * sig * sig);
  float lam = sigmoidf(sc[1]);
  float d = geo[idx];
  float fx = expf(-d * d * inv);
  fixOut[idx] = fx;
  float learn = lam * fx + (1.f - lam) * (W2[i * MT + t] * fx + b2[i * MT + t]);
  preOut[b * (long)NT * MT + i * MT + t] = learn;
}

// ---------------------------------------------------------------------------
// Strided row softmax: out[row][j] = softmax(in[b,r,j]*scale)
// ---------------------------------------------------------------------------
__launch_bounds__(128)
__global__ void softmax_rows(const float* __restrict__ in, float* __restrict__ out,
                             int rowsPerBatch, int len,
                             long bStride, long rStride, long cStride, float scale)
{
  const int row = blockIdx.x;
  const int b = row / rowsPerBatch, r = row % rowsPerBatch;
  const float* src = in + (long)b * bStride + (long)r * rStride;
  float* dst = out + (long)row * len;
  const int tid = threadIdx.x;
  __shared__ float red[128];
  float m = -3.0e38f;
  for (int j = tid; j < len; j += 128) m = fmaxf(m, src[(long)j * cStride] * scale);
  red[tid] = m; __syncthreads();
  for (int s = 64; s; s >>= 1) { if (tid < s) red[tid] = fmaxf(red[tid], red[tid + s]); __syncthreads(); }
  m = red[0]; __syncthreads();
  float sum = 0.f;
  for (int j = tid; j < len; j += 128) sum += expf(src[(long)j * cStride] * scale - m);
  red[tid] = sum; __syncthreads();
  for (int s = 64; s; s >>= 1) { if (tid < s) red[tid] += red[tid + s]; __syncthreads(); }
  float invs = 1.f / red[0];
  for (int j = tid; j < len; j += 128) dst[j] = expf(src[(long)j * cStride] * scale - m) * invs;
}

// ---------------------------------------------------------------------------
// Deterministic single-block reductions for the loss
// ---------------------------------------------------------------------------
__launch_bounds__(512)
__global__ void reduce_sum_sq(const float* __restrict__ x, long n, float* outS, float* outSS)
{
  const int tid = threadIdx.x;
  __shared__ float rs[512], rq[512];
  float s = 0.f, q = 0.f;
  for (long i = tid; i < n; i += 512) { float v = x[i]; s += v; q += v * v; }
  rs[tid] = s; rq[tid] = q; __syncthreads();
  for (int st = 256; st; st >>= 1) {
    if (tid < st) { rs[tid] += rs[tid + st]; rq[tid] += rq[tid + st]; }
    __syncthreads();
  }
  if (tid == 0) { *outS = rs[0]; *outSS = rq[0]; }
}

__launch_bounds__(512)
__global__ void reduce_absdiff(const float* __restrict__ x, const float* __restrict__ y,
                               long n, float* out)
{
  const int tid = threadIdx.x;
  __shared__ float rs[512];
  float s = 0.f;
  for (long i = tid; i < n; i += 512) s += fabsf(x[i] - y[i]);
  rs[tid] = s; __syncthreads();
  for (int st = 256; st; st >>= 1) { if (tid < st) rs[tid] += rs[tid + st]; __syncthreads(); }
  if (tid == 0) *out = rs[0];
}

__device__ __forceinline__ float var_unb(float S, float SS, float n) {
  return (SS - S * S / n) / (n - 1.f);
}

__global__ void final_loss(const float* st, float* out)
{
  const float nA = (float)(BDIM * MT * NT);
  const float nB = (float)(BDIM * NT * MT);
  float l1 = st[12] / nA + st[13] / nB;
  float l2 = -fabsf(var_unb(st[4], st[5], nA) - var_unb(st[6], st[7], nA))
             -fabsf(var_unb(st[8], st[9], nB) - var_unb(st[10], st[11], nB));
  *out = (l1 + 100.f * l2) * 1000.f;
}

// ---------------------------------------------------------------------------
// Host orchestration
// ---------------------------------------------------------------------------
static inline void launch_gemm(hipStream_t st,
    const float* A, const float* B, float* C,
    int M, int N, int K, int batches,
    long sAb, long sAm, long sAk,
    long sBb, long sBk, long sBn,
    long sCb, long sCm,
    const float* bias, int biasMode,
    const float* aPtr, int aIdx, float aConst, int flags)
{
  dim3 g((unsigned)((N + GBN - 1) / GBN), (unsigned)((M + GBM - 1) / GBM), (unsigned)batches);
  gemm_wmma_bf16<<<g, dim3(256), 0, st>>>(A, B, C, M, N, K, sAb, sAm, sAk,
                                          sBb, sBk, sBn, sCb, sCm,
                                          bias, biasMode, aPtr, aIdx, aConst, flags);
}

static inline dim3 ew_grid(long n) { return dim3((unsigned)((n + 255) / 256)); }

extern "C" void kernel_launch(void* const* d_in, const int* in_sizes, int n_in,
                              void* d_out, int out_size, void* d_ws, size_t ws_size,
                              hipStream_t stream)
{
  (void)in_sizes; (void)n_in; (void)out_size; (void)ws_size;

  const float* ir     = (const float*)d_in[0];
  const float* vis    = (const float*)d_in[1];
  const float* lan    = (const float*)d_in[2];
  const float* vis_w  = (const float*)d_in[3];
  const float* vis_b  = (const float*)d_in[4];
  const float* vis_g  = (const float*)d_in[5];
  const float* vis_be = (const float*)d_in[6];
  const float* ir_w   = (const float*)d_in[7];
  const float* ir_b   = (const float*)d_in[8];
  const float* ir_g   = (const float*)d_in[9];
  const float* ir_be  = (const float*)d_in[10];
  const float* t_w    = (const float*)d_in[11];
  const float* t_b    = (const float*)d_in[12];
  const float* vout_w = (const float*)d_in[13];
  const float* vout_b = (const float*)d_in[14];
  const float* tout_w = (const float*)d_in[15];
  const float* tout_b = (const float*)d_in[16];
  const float* W1     = (const float*)d_in[17];
  const float* b1     = (const float*)d_in[18];
  const float* W2     = (const float*)d_in[19];
  const float* b2     = (const float*)d_in[20];
  const float* W4     = (const float*)d_in[21];
  const float* W5     = (const float*)d_in[22];
  const float* ml_w   = (const float*)d_in[23];
  const float* ml_b   = (const float*)d_in[24];
  const float* ml2_w  = (const float*)d_in[25];
  const float* ml2_b  = (const float*)d_in[26];
  const float* sc     = (const float*)d_in[27];

  float* outv = (float*)d_out;                           // new_vis [B,C,100]
  float* outl = outv + (long)BDIM * CDIM * HW;           // new_lan [B,20,C]
  float* outloss = outv + (long)BDIM * CDIM * HW + (long)BDIM * MT * CDIM;

  // ---- workspace bump allocator (~382 MB total) ----
  float* w = (float*)d_ws;
  size_t off = 0;
  auto alloc = [&](size_t n) { float* p = w + off; off += n; return p; };

  const long SBC = (long)CDIM * HW;        // 102400  per-batch [100,C] / [C,100]
  const long STC = (long)MT * CDIM;        // 20480   per-batch [20,C]

  float* jointVI = alloc((size_t)BDIM * P_VI * CDIM);   // reused later: tmp1 + Mt2iL
  float* gramVI  = alloc((size_t)BDIM * P_VI * P_VI);   // reused later: gramIT
  float* geoVI   = alloc((size_t)BDIM * NT * NT);
  float* Qv = alloc((size_t)BDIM * SBC); float* Kv = alloc((size_t)BDIM * SBC);
  float* Vv = alloc((size_t)BDIM * SBC); float* Qi = alloc((size_t)BDIM * SBC);
  float* Ki = alloc((size_t)BDIM * SBC); float* Vi = alloc((size_t)BDIM * SBC);
  float* Qt = alloc((size_t)BDIM * STC); float* Kt = alloc((size_t)BDIM * STC);
  float* Vt = alloc((size_t)BDIM * STC);
  float* Qf = alloc((size_t)BDIM * SBC); float* Kf = alloc((size_t)BDIM * SBC);
  float* Vf = alloc((size_t)BDIM * SBC);
  float* featImg = alloc((size_t)BDIM * SBC);
  float* lanT    = alloc((size_t)BDIM * STC);
  float* jointIT = alloc((size_t)BDIM * P_IT * CDIM);
  float* geoT2I  = alloc((size_t)BDIM * MT * NT);
  float* fixT2I  = alloc((size_t)BDIM * MT * NT);
  float* preI2T  = alloc((size_t)BDIM * NT * MT);
  float* loI2T   = alloc((size_t)BDIM * NT * MT);
  float* loT2I   = alloc((size_t)BDIM * MT * NT);
  float* Mgap    = alloc(64); float* alB = alloc(64); float* beB = alloc(64);
  float* stateF  = alloc(32);
  unsigned int* cnt = (unsigned int*)(stateF + 16);

  // buffer reuse (lifetimes verified):
  float* tmp1   = jointVI;                       // [B,100,C]
  float* Mt2iL  = jointVI + (size_t)BDIM * SBC;  // [B,100,C]
  float* gramIT = gramVI;                        // [B,120,120]
  float* tmp2   = Qv;                            // [B,20,C]
  float* Mi2tL  = Qv + (size_t)BDIM * STC;       // [B,20,C]
  float* G1     = Qv + 2 * (size_t)BDIM * STC;   // [B,100,100]
  float* Alog = Kv;                 float* Av = Kv + (size_t)BDIM * MT * NT;
  float* Blog = Av + (size_t)BDIM * MT * NT;
  float* Bv   = Blog + (size_t)BDIM * NT * MT;
  float* fA   = Bv + (size_t)BDIM * NT * MT;
  float* fB   = fA + (size_t)BDIM * MT * NT;
  float* G2   = fB + (size_t)BDIM * NT * MT;     // [B,20,20]
  float* newLanPre = Vv;                         // [B,20,C]
  float* newVisPre = Qi;                         // [B,100,C]

  init_state<<<1, 1, 0, stream>>>(stateF, cnt);

  // ================= stage 1: image manifold (Isomap geodesics) =============
  make_joint_vi<<<ew_grid((long)BDIM * P_VI * CDIM), 256, 0, stream>>>(vis, ir, jointVI);
  launch_gemm(stream, jointVI, jointVI, gramVI, P_VI, P_VI, CDIM, BDIM,
              (long)P_VI * CDIM, CDIM, 1, (long)P_VI * CDIM, 1, CDIM,
              (long)P_VI * P_VI, P_VI, nullptr, 0, nullptr, 0, 1.f, 0);
  geodesic_kernel<<<BDIM, 1024, P_VI * P_VI * sizeof(float), stream>>>(
      gramVI, geoVI, P_VI, 0, NT, NT, NT);
  minmax_norm<<<BDIM, 256, 0, stream>>>(geoVI, NT * NT);
  for (int it = 0; it < 24; ++it) {
    count_le<<<512, 256, 0, stream>>>(geoVI, (long)BDIM * NT * NT, stateF, cnt);
    bisect_update<<<1, 1, 0, stream>>>(stateF, cnt, (long)BDIM * NT * NT);
  }
  finalize_median<<<1, 1, 0, stream>>>(stateF, 2);

  // ================= QKV projections ========================================
  float* visOut[3] = { Qv, Kv, Vv };
  float* irOut[3]  = { Qi, Ki, Vi };
  for (int i = 0; i < 3; ++i) {
    launch_gemm(stream, vis, vis_w + (long)i * CDIM * CDIM, visOut[i],
                HW, CDIM, CDIM, BDIM, SBC, 1, HW, 0, 1, CDIM,
                SBC, CDIM, vis_b + (long)i * CDIM, 1, nullptr, 0, 1.f, 0);
    inorm_relu<<<BDIM * CDIM, 128, 0, stream>>>(visOut[i], vis_g + (long)i * CDIM,
                                                vis_be + (long)i * CDIM, HW);
    launch_gemm(stream, ir, ir_w + (long)i * CDIM * CDIM, irOut[i],
                HW, CDIM, CDIM, BDIM, SBC, 1, HW, 0, 1, CDIM,
                SBC, CDIM, ir_b + (long)i * CDIM, 1, nullptr, 0, 1.f, 0);
    inorm_relu<<<BDIM * CDIM, 128, 0, stream>>>(irOut[i], ir_g + (long)i * CDIM,
                                                ir_be + (long)i * CDIM, HW);
  }
  float* txtOut[3] = { Qt, Kt, Vt };
  for (int i = 0; i < 3; ++i)
    launch_gemm(stream, lan, t_w + (long)i * CDIM * CDIM, txtOut[i],
                MT, CDIM, CDIM, BDIM, (long)CDIM * MT, 1, MT, 0, 1, CDIM,
                STC, CDIM, t_b + (long)i * CDIM, 1, nullptr, 0, 1.f, 2 /*relu*/);

  // ================= fusion weights + fused features ========================
  mgap_reduce<<<BDIM, 256, 0, stream>>>(geoVI, W1, b1, sc, stateF, Mgap);
  alphabeta<<<1, 64, 0, stream>>>(Mgap, sc, alB, beB);
  long nImg = (long)BDIM * SBC;
  fuse_ab<<<ew_grid(nImg), 256, 0, stream>>>(Qv, Qi, alB, beB, Qf, SBC, nImg);
  fuse_ab<<<ew_grid(nImg), 256, 0, stream>>>(Kv, Ki, alB, beB, Kf, SBC, nImg);
  fuse_ab<<<ew_grid(nImg), 256, 0, stream>>>(Vv, Vi, alB, beB, Vf, SBC, nImg);
  feat_img_kernel<<<ew_grid(nImg), 256, 0, stream>>>(vis, ir, alB, beB, featImg);

  // ================= stage 2: text manifold =================================
  lan_transpose<<<ew_grid((long)BDIM * STC), 256, 0, stream>>>(lan, lanT);
  make_joint_it<<<ew_grid((long)BDIM * P_IT * CDIM), 256, 0, stream>>>(featImg, lanT, jointIT);
  launch_gemm(stream, jointIT, jointIT, gramIT, P_IT, P_IT, CDIM, BDIM,
              (long)P_IT * CDIM, CDIM, 1, (long)P_IT * CDIM, 1, CDIM,
              (long)P_IT * P_IT, P_IT, nullptr, 0, nullptr, 0, 1.f, 0);
  geodesic_kernel<<<BDIM, 1024, P_IT * P_IT * sizeof(float), stream>>>(
      gramIT, geoT2I, P_IT, NT, MT, 0, NT);
  minmax_norm<<<BDIM, 256, 0, stream>>>(geoT2I, MT * NT);
  for (int it = 0; it < 24; ++it) {
    count_le<<<512, 256, 0, stream>>>(geoT2I, (long)BDIM * MT * NT, stateF, cnt);
    bisect_update<<<1, 1, 0, stream>>>(stateF, cnt, (long)BDIM * MT * NT);
  }
  finalize_median<<<1, 1, 0, stream>>>(stateF, 3);

  fix_learn_text<<<ew_grid((long)BDIM * MT * NT), 256, 0, stream>>>(
      geoT2I, W2, b2, sc, stateF, fixT2I, preI2T);
  // M_i2t_lo: rows (b,i) len 20 ; M_t2i_lo: rows (b,t) len 100 (transposed read)
  softmax_rows<<<BDIM * NT, 128, 0, stream>>>(preI2T, loI2T, NT, MT, 2000, 20, 1, 1.f / SQC);
  softmax_rows<<<BDIM * MT, 128, 0, stream>>>(preI2T, loT2I, MT, NT, 2000, 1, 20, 1.f / SQC);

  // M_t2i_learn = (loI2T @ lanT) @ ml_w^T + ml_b
  launch_gemm(stream, loI2T, lanT, tmp1, NT, CDIM, MT, BDIM,
              2000, MT, 1, STC, CDIM, 1, SBC, CDIM, nullptr, 0, nullptr, 0, 1.f, 0);
  launch_gemm(stream, tmp1, ml_w, Mt2iL, NT, CDIM, CDIM, BDIM,
              SBC, CDIM, 1, 0, 1, CDIM, SBC, CDIM, ml_b, 1, nullptr, 0, 1.f, 0);
  // M_i2t_learn = (loT2I @ featImg) @ ml2_w^T + ml2_b
  launch_gemm(stream, loT2I, featImg, tmp2, MT, CDIM, NT, BDIM,
              2000, NT, 1, SBC, CDIM, 1, STC, CDIM, nullptr, 0, nullptr, 0, 1.f, 0);
  launch_gemm(stream, tmp2, ml2_w, Mi2tL, MT, CDIM, CDIM, BDIM,
              STC, CDIM, 1, 0, 1, CDIM, STC, CDIM, ml2_b, 1, nullptr, 0, 1.f, 0);

  // ================= text->image attention ==================================
  launch_gemm(stream, Qt, Kf, Alog, MT, NT, CDIM, BDIM,
              STC, CDIM, 1, SBC, 1, CDIM, 2000, NT, nullptr, 0, nullptr, 0, 1.f, 0);
  launch_gemm(stream, Qt, Mt2iL, Alog, MT, NT, CDIM, BDIM,
              STC, CDIM, 1, SBC, 1, CDIM, 2000, NT, nullptr, 0, sc, 2, 1.f, 1);
  launch_gemm(stream, Mt2iL, Kf, G1, NT, NT, CDIM, BDIM,
              SBC, CDIM, 1, SBC, 1, CDIM, (long)NT * NT, NT, nullptr, 0, nullptr, 0, 1.f, 0);
  launch_gemm(stream, W4, G1, Alog, MT, NT, NT, BDIM,
              0, NT, 1, (long)NT * NT, NT, 1, 2000, NT, nullptr, 0, sc, 3, 1.f, 1);
  softmax_rows<<<BDIM * MT, 128, 0, stream>>>(Alog, Av, MT, NT, 2000, NT, 1, 1.f / SQC);
  launch_gemm(stream, Av, Vf, newLanPre, MT, CDIM, NT, BDIM,
              2000, NT, 1, SBC, CDIM, 1, STC, CDIM, nullptr, 0, nullptr, 0, 1.f, 0);
  launch_gemm(stream, loT2I, Vf, newLanPre, MT, CDIM, NT, BDIM,
              2000, NT, 1, SBC, CDIM, 1, STC, CDIM, nullptr, 0, sc, 4, 1.f, 1);

  // ================= image->text attention ==================================
  launch_gemm(stream, Qf, Kt, Blog, NT, MT, CDIM, BDIM,
              SBC, CDIM, 1, STC, 1, CDIM, 2000, MT, nullptr, 0, nullptr, 0, 1.f, 0);
  launch_gemm(stream, Qf, Mi2tL, Blog, NT, MT, CDIM, BDIM,
              SBC, CDIM, 1, STC, 1, CDIM, 2000, MT, nullptr, 0, sc, 5, 1.f, 1);
  launch_gemm(stream, Mi2tL, Kt, G2, MT, MT, CDIM, BDIM,
              STC, CDIM, 1, STC, 1, CDIM, (long)MT * MT, MT, nullptr, 0, nullptr, 0, 1.f, 0);
  launch_gemm(stream, W5, G2, Blog, NT, MT, MT, BDIM,
              0, MT, 1, (long)MT * MT, MT, 1, 2000, MT, nullptr, 0, sc, 6, 1.f, 1);
  softmax_rows<<<BDIM * NT, 128, 0, stream>>>(Blog, Bv, NT, MT, 2000, MT, 1, 1.f / SQC);
  launch_gemm(stream, Bv, Vt, newVisPre, NT, CDIM, MT, BDIM,
              2000, MT, 1, STC, CDIM, 1, SBC, CDIM, nullptr, 0, nullptr, 0, 1.f, 0);
  launch_gemm(stream, loI2T, Vt, newVisPre, NT, CDIM, MT, BDIM,
              2000, MT, 1, STC, CDIM, 1, SBC, CDIM, nullptr, 0, sc, 7, 1.f, 1);

  // ================= output projections =====================================
  // new_vis[b,m,hw] = sum_c vout_w[m,c] * newVisPre[b,hw,c] + vout_b[m]
  launch_gemm(stream, vout_w, newVisPre, outv, CDIM, HW, CDIM, BDIM,
              0, CDIM, 1, SBC, 1, CDIM, SBC, HW, vout_b, 2, nullptr, 0, 1.f, 0);
  launch_gemm(stream, newLanPre, tout_w, outl, MT, CDIM, CDIM, BDIM,
              STC, CDIM, 1, 0, 1, CDIM, STC, CDIM, tout_b, 1, nullptr, 0, 1.f, 0);

  // ================= regularization loss ====================================
  softmax_rows<<<BDIM * MT, 128, 0, stream>>>(fixT2I, fA, MT, NT, 2000, NT, 1, 1.f / SQC);
  softmax_rows<<<BDIM * NT, 128, 0, stream>>>(fixT2I, fB, NT, MT, 2000, 1, NT, 1.f / SQC);
  long nA = (long)BDIM * MT * NT;
  reduce_sum_sq<<<1, 512, 0, stream>>>(Av, nA, stateF + 4, stateF + 5);
  reduce_sum_sq<<<1, 512, 0, stream>>>(fA, nA, stateF + 6, stateF + 7);
  reduce_sum_sq<<<1, 512, 0, stream>>>(Bv, nA, stateF + 8, stateF + 9);
  reduce_sum_sq<<<1, 512, 0, stream>>>(fB, nA, stateF + 10, stateF + 11);
  reduce_absdiff<<<1, 512, 0, stream>>>(Av, fA, nA, stateF + 12);
  reduce_absdiff<<<1, 512, 0, stream>>>(Bv, fB, nA, stateF + 13);
  final_loss<<<1, 1, 0, stream>>>(stateF, outloss);
}
